// Attn_24232205484181
// MI455X (gfx1250) — matrix operand order repaired and checked by reading, not compiled, not run on hardware
//
#include <hip/hip_runtime.h>
#include <hip/hip_bf16.h>

// ---------------------------------------------------------------------------
// Types for CDNA5 WMMA (gfx1250, wave32)
// ---------------------------------------------------------------------------
typedef __attribute__((ext_vector_type(16))) __bf16 v16bf;
typedef __attribute__((ext_vector_type(8)))  float  v8f;
typedef __attribute__((ext_vector_type(4)))  unsigned int v4u;
typedef __attribute__((ext_vector_type(8)))  int v8i;
typedef __attribute__((ext_vector_type(4)))  int v4i;

union Frag {            // 16 bf16 = 8 VGPRs, loadable as two 16-byte chunks
    v16bf v;
    uint4 q[2];
};

static __device__ __forceinline__ v8f wmma_bf16(const Frag& a, const Frag& b, v8f c) {
    return __builtin_amdgcn_wmma_f32_16x16x32_bf16(
        false, a.v, false, b.v, (short)0, c, false, false);
}

static __device__ __forceinline__ unsigned short f2bf(float f) {
    union { float f; unsigned u; } x; x.f = f;
    unsigned r = x.u + 0x7fffu + ((x.u >> 16) & 1u);   // round-to-nearest-even
    return (unsigned short)(r >> 16);
}

#define DM     1024
#define NCTX   2048
#define BATCH  2
#define HEADS  16
#define DH     64
#define MTOT   (BATCH * NCTX)

#if defined(__has_builtin)
#if __has_builtin(__builtin_amdgcn_tensor_load_to_lds) && \
    __has_builtin(__builtin_amdgcn_s_wait_tensorcnt)
#define USE_TDM 1
#endif
#endif
#ifndef USE_TDM
#define USE_TDM 0
#endif

#if USE_TDM
// ---------------------------------------------------------------------------
// Issue a TDM 2-D tile load: tile = tile_rows x 32 bf16, row-major source with
// row stride 1024 elements; LDS rows padded by 16B (pad_interval 16 DWORDs,
// pad_amount 4 DWORDs) -> 40-ushort LDS row stride.
// (6-arg builtin form: g0, g1, g2, g3, g4-spare, cpol)
// ---------------------------------------------------------------------------
static __device__ __forceinline__ void tdm_load_tile32(const unsigned short* gptr,
                                                       unsigned lds_off,
                                                       unsigned tile_rows,
                                                       unsigned tensor_rows) {
    unsigned long long ga = (unsigned long long)(uintptr_t)gptr;
    v4u g0 = { 1u,                                   // count=1, user descriptor
               lds_off,                              // lds_addr (bytes)
               (unsigned)(ga & 0xffffffffu),         // global_addr lo
               (unsigned)(ga >> 32) | 0x80000000u }; // global_addr hi | type=2
    v8i g1 = { (int)((1u << 16)        // data_size = 2 bytes
                   | (1u << 20)        // pad_enable
                   | (3u << 22)        // pad_interval: 16 DWORDs (one 64B row)
                   | (3u << 25)),      // pad_amount: 4 DWORDs (16B)
               (int)(1024u << 16),                       // tensor_dim0 = 1024
               (int)((tensor_rows & 0xffffu) << 16),     // tensor_dim1 lo16
               (int)(32u << 16),                         // dim1 hi16=0 | tile_dim0=32
               (int)tile_rows,                           // tile_dim1 (tile_dim2=0)
               1024,                                     // tensor_dim0_stride lo32
               0, 0 };
    v4i z4 = { 0, 0, 0, 0 };
    v8i z8 = { 0, 0, 0, 0, 0, 0, 0, 0 };
    __builtin_amdgcn_tensor_load_to_lds(g0, g1, z4, z4, z8, 0);
}
#endif

// ---------------------------------------------------------------------------
// Kernel 1a: f32 -> bf16 row-major convert (x)
// ---------------------------------------------------------------------------
__global__ __launch_bounds__(256) void cvt_bf16(const float* __restrict__ src,
                                                unsigned short* __restrict__ dst,
                                                int n4) {
    int i = blockIdx.x * blockDim.x + threadIdx.x;
    if (i < n4) {
        float4 f = ((const float4*)src)[i];
        ushort4 o;
        o.x = f2bf(f.x); o.y = f2bf(f.y); o.z = f2bf(f.z); o.w = f2bf(f.w);
        ((ushort4*)dst)[i] = o;
    }
}

// ---------------------------------------------------------------------------
// Kernel 1b: f32 [k][n] -> bf16 transposed [n][k] (weights). 32x32 LDS tiles.
// ---------------------------------------------------------------------------
__global__ __launch_bounds__(256) void cvt_tr_bf16(const float* __restrict__ src,
                                                   unsigned short* __restrict__ dst) {
    __shared__ float t[32][33];
    const int n0 = blockIdx.x * 32, k0 = blockIdx.y * 32;
    const int tx = threadIdx.x & 31, ty = threadIdx.x >> 5;   // ty: 0..7
#pragma unroll
    for (int i = ty; i < 32; i += 8) t[i][tx] = src[(size_t)(k0 + i) * DM + n0 + tx];
    __syncthreads();
#pragma unroll
    for (int i = ty; i < 32; i += 8)
        dst[(size_t)(n0 + i) * DM + k0 + tx] = f2bf(t[tx][i]);
}

// ---------------------------------------------------------------------------
// Kernel 2/4: bf16 WMMA GEMM, C[4096,1024] = A[4096,1024] * Bt^T, where
// Bt is the TRANSPOSED weight [n][k] bf16.
//   MODE 0: bf16 out, per-head layout [(b*16+h)][n][64], (acc+bias)*scale  (Q,K)
//   MODE 2: bf16 out, transposed head layout [(b*16+h)*64+d][n], acc+bias  (V)
//   MODE 1: f32 out row-major [M,1024], no bias                            (out)
// Block 256 threads (8 waves); tile 128x64; K-step 32; TDM double-buffered.
// ---------------------------------------------------------------------------
template <int MODE>
__global__ __launch_bounds__(256) void gemm_bf16(const unsigned short* __restrict__ A,
                                                 const unsigned short* __restrict__ Bt,
                                                 const float* __restrict__ bias,
                                                 void* __restrict__ dst,
                                                 float scale) {
    const int tid  = threadIdx.x;
    const int wave = tid >> 5;
    const int lane = tid & 31;
    const int mbase = blockIdx.y * 128;
    const int nbase = blockIdx.x * 64;
    const int wm = wave >> 1;
    const int wn = wave & 1;
    const int sel = lane >> 4;
    const int lm  = lane & 15;

    __shared__ __align__(16) unsigned short Abuf[2][128][40];  // [m][k]
    __shared__ __align__(16) unsigned short Bbuf[2][64][40];   // [n][k]

    v8f acc[2][2];
    const v8f vzero = {0.f, 0.f, 0.f, 0.f, 0.f, 0.f, 0.f, 0.f};
#pragma unroll
    for (int i = 0; i < 2; i++)
#pragma unroll
        for (int j = 0; j < 2; j++) acc[i][j] = vzero;

#if USE_TDM
    const unsigned ldsA0 = (unsigned)(uintptr_t)(void*)&Abuf[0][0][0];
    const unsigned ldsA1 = (unsigned)(uintptr_t)(void*)&Abuf[1][0][0];
    const unsigned ldsB0 = (unsigned)(uintptr_t)(void*)&Bbuf[0][0][0];
    const unsigned ldsB1 = (unsigned)(uintptr_t)(void*)&Bbuf[1][0][0];
    if (wave == 0) {
        tdm_load_tile32(A  + (size_t)mbase * DM, ldsA0, 128u, (unsigned)MTOT);
        tdm_load_tile32(Bt + (size_t)nbase * DM, ldsB0, 64u,  (unsigned)DM);
    }
    int cur = 0;
    for (int kb = 0; kb < DM; kb += 32) {
        if (wave == 0) __builtin_amdgcn_s_wait_tensorcnt(0);
        __syncthreads();
        if (wave == 0 && kb + 32 < DM) {
            tdm_load_tile32(A  + (size_t)mbase * DM + kb + 32,
                            cur ? ldsA0 : ldsA1, 128u, (unsigned)MTOT);
            tdm_load_tile32(Bt + (size_t)nbase * DM + kb + 32,
                            cur ? ldsB0 : ldsB1, 64u, (unsigned)DM);
        }
        Frag afr[2], bfr[2];
#pragma unroll
        for (int mt = 0; mt < 2; mt++) {
            int mr = wm * 32 + mt * 16 + lm;
            afr[mt].q[0] = *(const uint4*)&Abuf[cur][mr][sel * 8];
            afr[mt].q[1] = *(const uint4*)&Abuf[cur][mr][16 + sel * 8];
        }
#pragma unroll
        for (int nt = 0; nt < 2; nt++) {
            int nc = wn * 32 + nt * 16 + lm;
            bfr[nt].q[0] = *(const uint4*)&Bbuf[cur][nc][sel * 8];
            bfr[nt].q[1] = *(const uint4*)&Bbuf[cur][nc][16 + sel * 8];
        }
#pragma unroll
        for (int mt = 0; mt < 2; mt++)
#pragma unroll
            for (int nt = 0; nt < 2; nt++)
                acc[mt][nt] = wmma_bf16(afr[mt], bfr[nt], acc[mt][nt]);
        cur ^= 1;
    }
#else
    // Fallback: manual staging (contiguous b128 both ways, Bt pre-transposed)
    for (int kb = 0; kb < DM; kb += 32) {
        __syncthreads();
        {   // A tile 128x32
            int row = tid >> 1, cb = (tid & 1) * 16;
            const uint4* g = (const uint4*)(A + (size_t)(mbase + row) * DM + kb + cb);
            *(uint4*)&Abuf[0][row][cb]     = g[0];
            *(uint4*)&Abuf[0][row][cb + 8] = g[1];
        }
        {   // B tile 64x32 (already [n][k])
            int n = tid >> 2, cb = (tid & 3) * 8;
            *(uint4*)&Bbuf[0][n][cb] =
                *(const uint4*)(Bt + (size_t)(nbase + n) * DM + kb + cb);
        }
        __syncthreads();
        Frag afr[2], bfr[2];
#pragma unroll
        for (int mt = 0; mt < 2; mt++) {
            int mr = wm * 32 + mt * 16 + lm;
            afr[mt].q[0] = *(const uint4*)&Abuf[0][mr][sel * 8];
            afr[mt].q[1] = *(const uint4*)&Abuf[0][mr][16 + sel * 8];
        }
#pragma unroll
        for (int nt = 0; nt < 2; nt++) {
            int nc = wn * 32 + nt * 16 + lm;
            bfr[nt].q[0] = *(const uint4*)&Bbuf[0][nc][sel * 8];
            bfr[nt].q[1] = *(const uint4*)&Bbuf[0][nc][16 + sel * 8];
        }
#pragma unroll
        for (int mt = 0; mt < 2; mt++)
#pragma unroll
            for (int nt = 0; nt < 2; nt++)
                acc[mt][nt] = wmma_bf16(afr[mt], bfr[nt], acc[mt][nt]);
    }
#endif

    // ---- epilogue ----
#pragma unroll
    for (int mt = 0; mt < 2; mt++) {
#pragma unroll
        for (int nt = 0; nt < 2; nt++) {
            int mg0 = mbase + wm * 32 + mt * 16 + 8 * sel;
            int cg  = nbase + wn * 32 + nt * 16 + lm;
            if (MODE == 1) {
#pragma unroll
                for (int r = 0; r < 8; r++)
                    ((float*)dst)[(size_t)(mg0 + r) * DM + cg] = acc[mt][nt][r];
            } else {
                float bv = bias[cg];
                int h = cg >> 6, d = cg & 63;
#pragma unroll
                for (int r = 0; r < 8; r++) {
                    int m  = mg0 + r;
                    int b  = m >> 11;
                    int nc = m & 2047;
                    unsigned short val = f2bf((acc[mt][nt][r] + bv) * scale);
                    if (MODE == 0)   // Q/K: [(b*16+h)][n][64]
                        ((unsigned short*)dst)[(size_t)(b * HEADS + h) * (NCTX * DH)
                                               + (size_t)nc * DH + d] = val;
                    else             // V: transposed [(b*16+h)*64+d][n]
                        ((unsigned short*)dst)[((size_t)(b * HEADS + h) * DH + d) * NCTX
                                               + nc] = val;
                }
            }
        }
    }
}

// ---------------------------------------------------------------------------
// Kernel 3: flash attention, half-causal mask. No block barriers: Q/K frags
// come straight from global ([bh][n][64]); V frags straight from transposed
// global ([bh*64+d][n]); only P round-trips through per-wave LDS.
// Grid (NCTX/64, B*H), block 128 (4 waves, 16 query rows each).
// ---------------------------------------------------------------------------
__global__ __launch_bounds__(128) void flash_halfcausal(
        const unsigned short* __restrict__ Q,
        const unsigned short* __restrict__ K,
        const unsigned short* __restrict__ Vt,    // transposed
        unsigned short* __restrict__ Aout) {
    const int tid  = threadIdx.x;
    const int wave = tid >> 5;
    const int lane = tid & 31;
    const int sel  = lane >> 4;
    const int lm   = lane & 15;

    const int qb = blockIdx.x * 64;
    const int bh = blockIdx.y;
    const int qw = qb + wave * 16;
    const bool bottom = (qb >= NCTX / 2);
    const int kmax = bottom ? (qw + 16) : (NCTX / 2);   // per-wave causal bound

    const unsigned short* Qh = Q  + (size_t)bh * NCTX * DH;
    const unsigned short* Kh = K  + (size_t)bh * NCTX * DH;
    const unsigned short* Vh = Vt + (size_t)bh * DH * NCTX;   // [d][n]

    __shared__ __align__(16) unsigned short Pbuf[4][16][40];

    Frag qf[2];
#pragma unroll
    for (int dc = 0; dc < 2; dc++) {
        const unsigned short* qrow = Qh + (size_t)(qw + lm) * DH + dc * 32;
        qf[dc].q[0] = *(const uint4*)(qrow + sel * 8);
        qf[dc].q[1] = *(const uint4*)(qrow + 16 + sel * 8);
    }

    const v8f vzero = {0.f, 0.f, 0.f, 0.f, 0.f, 0.f, 0.f, 0.f};
    v8f o[4];
#pragma unroll
    for (int t = 0; t < 4; t++) o[t] = vzero;
    float mrun[8], lrun[8];
#pragma unroll
    for (int r = 0; r < 8; r++) { mrun[r] = -1e30f; lrun[r] = 0.f; }

    for (int kb = 0; kb < kmax; kb += 32) {
        // ---- S = (Q/sqrt(dk)) K^T ----
        v8f s[2];
        s[0] = vzero; s[1] = vzero;
#pragma unroll
        for (int nt = 0; nt < 2; nt++) {
#pragma unroll
            for (int dc = 0; dc < 2; dc++) {
                Frag kf;
                const unsigned short* krow =
                    Kh + (size_t)(kb + nt * 16 + lm) * DH + dc * 32 + sel * 8;
                kf.q[0] = *(const uint4*)(krow);
                kf.q[1] = *(const uint4*)(krow + 16);
                s[nt] = wmma_bf16(qf[dc], kf, s[nt]);
            }
        }

        // ---- mask (bottom half: j <= i) ----
        if (bottom) {
#pragma unroll
            for (int nt = 0; nt < 2; nt++) {
#pragma unroll
                for (int r = 0; r < 8; r++) {
                    int mg = qw + r + 8 * sel;
                    int ng = kb + nt * 16 + lm;
                    if (ng > mg) s[nt][r] = -1e30f;
                }
            }
        }

        // ---- online softmax (rows live in 16-lane halves) ----
#pragma unroll
        for (int r = 0; r < 8; r++) {
            float mx = fmaxf(s[0][r], s[1][r]);
            mx = fmaxf(mx, __shfl_xor(mx, 1, 32));
            mx = fmaxf(mx, __shfl_xor(mx, 2, 32));
            mx = fmaxf(mx, __shfl_xor(mx, 4, 32));
            mx = fmaxf(mx, __shfl_xor(mx, 8, 32));
            float mnew  = fmaxf(mrun[r], mx);
            float alpha = __expf(mrun[r] - mnew);
            float e0 = __expf(s[0][r] - mnew);
            float e1 = __expf(s[1][r] - mnew);
            float rs = e0 + e1;
            rs += __shfl_xor(rs, 1, 32);
            rs += __shfl_xor(rs, 2, 32);
            rs += __shfl_xor(rs, 4, 32);
            rs += __shfl_xor(rs, 8, 32);
            lrun[r] = lrun[r] * alpha + rs;
            mrun[r] = mnew;
#pragma unroll
            for (int t = 0; t < 4; t++) o[t][r] *= alpha;
            Pbuf[wave][r + 8 * sel][lm]      = f2bf(e0);
            Pbuf[wave][r + 8 * sel][16 + lm] = f2bf(e1);
        }
        asm volatile("s_wait_dscnt 0" ::: "memory");   // intra-wave P RAW

        Frag pf;
        pf.q[0] = *(const uint4*)&Pbuf[wave][lm][sel * 8];
        pf.q[1] = *(const uint4*)&Pbuf[wave][lm][16 + sel * 8];

        // ---- O += P * V (V frags straight from transposed global) ----
#pragma unroll
        for (int dt = 0; dt < 4; dt++) {
            Frag vf;
            const unsigned short* vrow =
                Vh + (size_t)(dt * 16 + lm) * NCTX + kb + sel * 8;
            vf.q[0] = *(const uint4*)(vrow);
            vf.q[1] = *(const uint4*)(vrow + 16);
            o[dt] = wmma_bf16(pf, vf, o[dt]);
        }
    }

    const int b = bh >> 4, h = bh & 15;
#pragma unroll
    for (int dt = 0; dt < 4; dt++) {
#pragma unroll
        for (int r = 0; r < 8; r++) {
            int row = qw + r + 8 * sel;
            int col = h * DH + dt * 16 + lm;
            Aout[(size_t)(b * NCTX + row) * DM + col] = f2bf(o[dt][r] / lrun[r]);
        }
    }
}

// ---------------------------------------------------------------------------
// Host launcher
// ---------------------------------------------------------------------------
extern "C" void kernel_launch(void* const* d_in, const int* in_sizes, int n_in,
                              void* d_out, int out_size, void* d_ws, size_t ws_size,
                              hipStream_t stream) {
    const float* x  = (const float*)d_in[0];
    const float* Wq = (const float*)d_in[1];
    const float* bq = (const float*)d_in[2];
    const float* Wk = (const float*)d_in[3];
    const float* bk = (const float*)d_in[4];
    const float* Wv = (const float*)d_in[5];
    const float* bv = (const float*)d_in[6];
    const float* Wo = (const float*)d_in[7];

    char* ws = (char*)d_ws;
    const size_t MB = 1024ull * 1024ull;
    unsigned short* xbf  = (unsigned short*)(ws + 0 * MB);    // 8 MB
    unsigned short* wqt  = (unsigned short*)(ws + 8 * MB);    // 2 MB  [n][k]
    unsigned short* wkt  = (unsigned short*)(ws + 10 * MB);
    unsigned short* wvt  = (unsigned short*)(ws + 12 * MB);
    unsigned short* wot  = (unsigned short*)(ws + 14 * MB);
    unsigned short* qws  = (unsigned short*)(ws + 16 * MB);   // 8 MB [bh][n][64]
    unsigned short* kws  = (unsigned short*)(ws + 24 * MB);
    unsigned short* vws  = (unsigned short*)(ws + 32 * MB);   // 8 MB [bh*64+d][n]
    unsigned short* aout = (unsigned short*)(ws + 40 * MB);   // 8 MB [m][1024]

    cvt_bf16<<<(MTOT * DM / 4 + 255) / 256, 256, 0, stream>>>(x, xbf, MTOT * DM / 4);
    dim3 gt(DM / 32, DM / 32);   // (32, 32)
    cvt_tr_bf16<<<gt, 256, 0, stream>>>(Wq, wqt);
    cvt_tr_bf16<<<gt, 256, 0, stream>>>(Wk, wkt);
    cvt_tr_bf16<<<gt, 256, 0, stream>>>(Wv, wvt);
    cvt_tr_bf16<<<gt, 256, 0, stream>>>(Wo, wot);

    dim3 gg(DM / 64, MTOT / 128);   // (16, 32)
    gemm_bf16<0><<<gg, 256, 0, stream>>>(xbf, wqt, bq, qws, 0.125f);
    gemm_bf16<0><<<gg, 256, 0, stream>>>(xbf, wkt, bk, kws, 1.0f);
    gemm_bf16<2><<<gg, 256, 0, stream>>>(xbf, wvt, bv, vws, 1.0f);

    flash_halfcausal<<<dim3(NCTX / 64, BATCH * HEADS), 128, 0, stream>>>(qws, kws, vws, aout);

    gemm_bf16<1><<<gg, 256, 0, stream>>>(aout, wot, nullptr, d_out, 1.0f);
}
